// MultiHeadAttention_54520314856024
// MI455X (gfx1250) — matrix operand order repaired and checked by reading, not compiled, not run on hardware
//
#include <hip/hip_runtime.h>

typedef __bf16 bf16_t;
typedef __attribute__((ext_vector_type(16))) __bf16 v16bf;
typedef __attribute__((ext_vector_type(8)))  float  v8f;
typedef int v4i __attribute__((vector_size(16)));

#define NB   8      // batch
#define NSQ  1024   // query len
#define NSK  1024   // key len
#define ND   1024   // states dim
#define NE   128    // head dim
#define NH   32     // heads

// ---------- CDNA5 async global->LDS path ----------
#if defined(__has_builtin)
#if __has_builtin(__builtin_amdgcn_global_load_async_to_lds_b128)
#define HAVE_ASYNC 1
#endif
#endif
#ifndef HAVE_ASYNC
#define HAVE_ASYNC 0
#endif

static __device__ __forceinline__ void async_copy16(const void* g, void* l) {
#if HAVE_ASYNC
  // ASYNCcnt-tracked DMA: 16B global -> LDS, no VGPR data movement.
  __builtin_amdgcn_global_load_async_to_lds_b128(
      (__attribute__((address_space(1))) v4i*)(unsigned long long)(size_t)g,
      (__attribute__((address_space(3))) v4i*)(unsigned)(size_t)l,
      0, 0);
#else
  *reinterpret_cast<uint4*>(l) = *reinterpret_cast<const uint4*>(g);
#endif
}

static __device__ __forceinline__ void wait_async0() {
#if HAVE_ASYNC
#if __has_builtin(__builtin_amdgcn_s_wait_asynccnt)
  __builtin_amdgcn_s_wait_asynccnt(0);
#else
  asm volatile("s_wait_asynccnt 0x0" ::: "memory");
#endif
#endif
}

// ---------- helpers ----------

static __device__ __forceinline__ bf16_t f2bf(float f) {
  unsigned u = __builtin_bit_cast(unsigned, f);
  unsigned r = u + 0x7FFFu + ((u >> 16) & 1u);   // round-to-nearest-even
  unsigned short h = (unsigned short)(r >> 16);
  return __builtin_bit_cast(bf16_t, h);
}

static __device__ __forceinline__ v8f wmma_bf16(v16bf a, v16bf b, v8f c) {
  return __builtin_amdgcn_wmma_f32_16x16x32_bf16(false, a, false, b, (short)0, c, false, false);
}

// A fragment (16x32 MxK, 16-bit): lane<16 -> row=lane, K {0..7,16..23};
// lane>=16 -> row=lane-16, K {8..15,24..31}. Storage: [row][k].
static __device__ __forceinline__ v16bf load_fragA(const bf16_t* t, int stride, int lane) {
  int row  = lane & 15;
  int koff = (lane >> 4) * 8;
  const bf16_t* p0 = t + row * stride + koff;
  const bf16_t* p1 = t + row * stride + koff + 16;
  v16bf r;
#pragma unroll
  for (int i = 0; i < 8; ++i) r[i] = p0[i];
#pragma unroll
  for (int i = 0; i < 8; ++i) r[8 + i] = p1[i];
  return r;
}

// B fragment (32x16 KxN, 16-bit): lane<16 -> col=lane, K {0..7,16..23};
// lane>=16 -> col=lane-16, K {8..15,24..31}. Storage: [col][k].
static __device__ __forceinline__ v16bf load_fragB(const bf16_t* t, int stride, int lane) {
  int col   = lane & 15;
  int kbase = (lane >> 4) * 8;
  const bf16_t* p = t + col * stride + kbase;
  v16bf r;
#pragma unroll
  for (int i = 0; i < 8; ++i) r[i] = p[i];
#pragma unroll
  for (int i = 0; i < 8; ++i) r[8 + i] = p[16 + i];
  return r;
}

// ---------- prep kernel A: states f32 -> bf16 ----------
__global__ __launch_bounds__(256) void cvt_states_kernel(
    const float* __restrict__ in, bf16_t* __restrict__ out) {
  size_t i = ((size_t)blockIdx.x * 256 + threadIdx.x) * 4;
  float4 v = *reinterpret_cast<const float4*>(in + i);
  out[i + 0] = f2bf(v.x);
  out[i + 1] = f2bf(v.y);
  out[i + 2] = f2bf(v.z);
  out[i + 3] = f2bf(v.w);
}

// ---------- prep kernel B: [batch][R][128] f32 -> [batch][128][R] bf16 ----------
__global__ __launch_bounds__(256) void tconv_kernel(
    const float* __restrict__ in, bf16_t* __restrict__ out, int R) {
  size_t base = (size_t)blockIdx.y * R * NE;
  int idx = blockIdx.x * 256 + threadIdx.x;
  int r = idx >> 7, c = idx & 127;
  out[base + (size_t)c * R + r] = f2bf(in[base + (size_t)r * NE + c]);
}

// ---------- kernel 1: fused K/V head projections ----------
// grid: (NSK/128, NH*NB), block: 256 (8 waves). Double-buffered async LDS staging.
__global__ __launch_bounds__(256) void kv_proj_kernel(
    const bf16_t* __restrict__ statesBf, const bf16_t* __restrict__ WkT,
    const bf16_t* __restrict__ WvT, const float* __restrict__ bk,
    const float* __restrict__ bv, bf16_t* __restrict__ keys,
    bf16_t* __restrict__ valsT) {
  __shared__ bf16_t As[2][128 * 40];
  __shared__ bf16_t Bk[2][128 * 40];
  __shared__ bf16_t Bv[2][128 * 40];

  const int hb = blockIdx.y;
  const int h  = hb / NB;
  const int b  = hb % NB;
  const int M0 = blockIdx.x * 128;
  const int t  = threadIdx.x;
  const int lane = t & 31, wm = t >> 5;

  const bf16_t* aG = statesBf + ((size_t)b * NSK + M0) * ND;
  const bf16_t* kG = WkT + (size_t)h * NE * ND;
  const bf16_t* vG = WvT + (size_t)h * NE * ND;

  auto stage = [&](int buf, int k0) {
#pragma unroll
    for (int j = 0; j < 2; ++j) {
      int id = t + j * 256;           // 0..511 -> (row, 16B chunk)
      int m = id >> 2, c = (id & 3) * 8;
      async_copy16(aG + (size_t)m * ND + k0 + c, &As[buf][m * 40 + c]);
      async_copy16(kG + (size_t)m * ND + k0 + c, &Bk[buf][m * 40 + c]);
      async_copy16(vG + (size_t)m * ND + k0 + c, &Bv[buf][m * 40 + c]);
    }
  };

  v8f accK[8], accV[8];
#pragma unroll
  for (int i = 0; i < 8; ++i) { accK[i] = (v8f){}; accV[i] = (v8f){}; }

  stage(0, 0);
  wait_async0();
  __syncthreads();

  const int NT = ND / 32;
  for (int kt = 0; kt < NT; ++kt) {
    const int cur = kt & 1;
    if (kt + 1 < NT) stage(cur ^ 1, (kt + 1) * 32);   // overlap DMA with WMMA

    const v16bf a = load_fragA(&As[cur][wm * 16 * 40], 40, lane);
    v16bf fb[8];
#pragma unroll
    for (int nt = 0; nt < 8; ++nt) fb[nt] = load_fragB(&Bk[cur][nt * 16 * 40], 40, lane);
#pragma unroll
    for (int nt = 0; nt < 8; ++nt) accK[nt] = wmma_bf16(a, fb[nt], accK[nt]);
#pragma unroll
    for (int nt = 0; nt < 8; ++nt) fb[nt] = load_fragB(&Bv[cur][nt * 16 * 40], 40, lane);
#pragma unroll
    for (int nt = 0; nt < 8; ++nt) accV[nt] = wmma_bf16(a, fb[nt], accV[nt]);

    if (kt + 1 < NT) { wait_async0(); __syncthreads(); }
  }

  const int colb  = lane & 15;
  const int rbase = (lane >> 4) * 8;
#pragma unroll
  for (int nt = 0; nt < 8; ++nt) {
    int col = nt * 16 + colb;
    float bkc = bk[h * NE + col];
    float bvc = bv[h * NE + col];
#pragma unroll
    for (int r = 0; r < 8; ++r) {
      int m = M0 + wm * 16 + rbase + r;
      keys [((size_t)hb * NSK + m) * NE + col] = f2bf(accK[nt][r] + bkc);
      valsT[((size_t)hb * NE + col) * NSK + m] = f2bf(accV[nt][r] + bvc);
    }
  }
}

// ---------- kernel 2: flash attention (one wave = one 16-query tile) ----------
// grid: (NSQ/16, NB, NH), block: 32.
__global__ __launch_bounds__(32) void flash_attn_kernel(
    const float* __restrict__ query, const bf16_t* __restrict__ keys,
    const bf16_t* __restrict__ valsT, bf16_t* __restrict__ ctx) {
  __shared__ bf16_t P[16 * 40];

  const int q0 = blockIdx.x * 16;
  const int b  = blockIdx.y;
  const int h  = blockIdx.z;
  const int hb = h * NB + b;
  const int lane = threadIdx.x;
  const int row  = lane & 15;
  const int koff = (lane >> 4) * 8;
  const float scale = 0.08838834764831845f;  // 1/sqrt(128)

  v16bf qf[4];
#pragma unroll
  for (int ec = 0; ec < 4; ++ec) {
    const float* qp = query + ((size_t)b * NSQ + q0 + row) * NE + ec * 32;
#pragma unroll
    for (int i = 0; i < 8; ++i) qf[ec][i] = f2bf(qp[koff + i]);
#pragma unroll
    for (int i = 0; i < 8; ++i) qf[ec][8 + i] = f2bf(qp[koff + 16 + i]);
  }

  v8f acc[8];
#pragma unroll
  for (int i = 0; i < 8; ++i) acc[i] = (v8f){};
  float mrow[8], lrow[8];
#pragma unroll
  for (int r = 0; r < 8; ++r) { mrow[r] = -3.0e38f; lrow[r] = 0.0f; }

  for (int k0 = 0; k0 < NSK; k0 += 32) {
    const bf16_t* kbase = keys + ((size_t)hb * NSK + k0) * NE;
    const bf16_t* vbase = valsT + (size_t)hb * NE * NSK + k0;
    if (k0 + 32 < NSK) {
      __builtin_prefetch(kbase + 32 * NE + row * NE);
      __builtin_prefetch(vbase + 32 + row * NSK);
    }
    // preload all 8 key fragments, then 8 score WMMAs
    v16bf kf[8];
#pragma unroll
    for (int ec = 0; ec < 4; ++ec) {
      kf[ec]     = load_fragB(kbase + ec * 32, NE, lane);
      kf[4 + ec] = load_fragB(kbase + 16 * NE + ec * 32, NE, lane);
    }
    v8f s0 = (v8f){}, s1 = (v8f){};
#pragma unroll
    for (int ec = 0; ec < 4; ++ec) {
      s0 = wmma_bf16(qf[ec], kf[ec], s0);
      s1 = wmma_bf16(qf[ec], kf[4 + ec], s1);
    }
    // issue V fragment loads now; softmax VALU work hides the latency
    v16bf vf[8];
#pragma unroll
    for (int nt = 0; nt < 8; ++nt)
      vf[nt] = load_fragB(vbase + (size_t)nt * 16 * NSK, NSK, lane);

    float corr[8];
#pragma unroll
    for (int r = 0; r < 8; ++r) {
      float x0 = s0[r] * scale, x1 = s1[r] * scale;
      float mx = fmaxf(x0, x1);
#pragma unroll
      for (int off = 1; off < 16; off <<= 1) mx = fmaxf(mx, __shfl_xor(mx, off, 16));
      float mnew = fmaxf(mrow[r], mx);
      float p0 = __expf(x0 - mnew), p1 = __expf(x1 - mnew);
      float c  = __expf(mrow[r] - mnew);
      float rs = p0 + p1;
#pragma unroll
      for (int off = 1; off < 16; off <<= 1) rs += __shfl_xor(rs, off, 16);
      lrow[r] = lrow[r] * c + rs;
      mrow[r] = mnew;
      corr[r] = c;
      int prow = r + koff;
      P[prow * 40 + row]      = f2bf(p0);
      P[prow * 40 + 16 + row] = f2bf(p1);
    }
#pragma unroll
    for (int nt = 0; nt < 8; ++nt)
#pragma unroll
      for (int r = 0; r < 8; ++r) acc[nt][r] *= corr[r];

    v16bf pa = load_fragA(P, 40, lane);
#pragma unroll
    for (int nt = 0; nt < 8; ++nt) acc[nt] = wmma_bf16(pa, vf[nt], acc[nt]);
  }

  float inv[8];
#pragma unroll
  for (int r = 0; r < 8; ++r) inv[r] = 1.0f / lrow[r];
#pragma unroll
  for (int nt = 0; nt < 8; ++nt)
#pragma unroll
    for (int r = 0; r < 8; ++r) {
      int qrow = q0 + r + koff;
      ctx[((size_t)b * NSQ + qrow) * (NH * NE) + h * NE + nt * 16 + row] =
          f2bf(acc[nt][r] * inv[r]);
    }
}

// ---------- kernel 3: out = ctx[8192x4096] x Wc[4096x128] + bc ----------
// grid: (NB*NSQ/128), block: 256 (8 waves). Double-buffered async LDS staging.
__global__ __launch_bounds__(256) void out_proj_kernel(
    const bf16_t* __restrict__ ctx, const bf16_t* __restrict__ WcT,
    const float* __restrict__ bc, float* __restrict__ out) {
  __shared__ bf16_t As[2][128 * 40];
  __shared__ bf16_t Bs[2][128 * 40];

  const int M0 = blockIdx.x * 128;
  const int t  = threadIdx.x;
  const int lane = t & 31, wm = t >> 5;
  const int K = NH * NE;  // 4096

  const bf16_t* aG = ctx + (size_t)M0 * K;

  auto stage = [&](int buf, int k0) {
#pragma unroll
    for (int j = 0; j < 2; ++j) {
      int id = t + j * 256;
      int m = id >> 2, c = (id & 3) * 8;
      async_copy16(aG + (size_t)m * K + k0 + c, &As[buf][m * 40 + c]);
      async_copy16(WcT + (size_t)m * K + k0 + c, &Bs[buf][m * 40 + c]);
    }
  };

  v8f acc[8];
#pragma unroll
  for (int i = 0; i < 8; ++i) acc[i] = (v8f){};

  stage(0, 0);
  wait_async0();
  __syncthreads();

  const int NT = K / 32;
  for (int kt = 0; kt < NT; ++kt) {
    const int cur = kt & 1;
    if (kt + 1 < NT) stage(cur ^ 1, (kt + 1) * 32);

    const v16bf a = load_fragA(&As[cur][wm * 16 * 40], 40, lane);
    v16bf fb[8];
#pragma unroll
    for (int nt = 0; nt < 8; ++nt) fb[nt] = load_fragB(&Bs[cur][nt * 16 * 40], 40, lane);
#pragma unroll
    for (int nt = 0; nt < 8; ++nt) acc[nt] = wmma_bf16(a, fb[nt], acc[nt]);

    if (kt + 1 < NT) { wait_async0(); __syncthreads(); }
  }

  const int colb  = lane & 15;
  const int rbase = (lane >> 4) * 8;
#pragma unroll
  for (int nt = 0; nt < 8; ++nt) {
    int col = nt * 16 + colb;
    float bias = bc[col];
#pragma unroll
    for (int r = 0; r < 8; ++r) {
      int m = M0 + wm * 16 + rbase + r;
      out[(size_t)m * NE + col] = acc[nt][r] + bias;
    }
  }
}

// ---------- launch ----------
extern "C" void kernel_launch(void* const* d_in, const int* in_sizes, int n_in,
                              void* d_out, int out_size, void* d_ws, size_t ws_size,
                              hipStream_t stream) {
  (void)in_sizes; (void)n_in; (void)out_size; (void)ws_size;
  const float* query  = (const float*)d_in[0];
  const float* states = (const float*)d_in[1];
  const float* Wk     = (const float*)d_in[2];
  const float* bk     = (const float*)d_in[3];
  const float* Wv     = (const float*)d_in[4];
  const float* bv     = (const float*)d_in[5];
  const float* Wc     = (const float*)d_in[6];
  const float* bc     = (const float*)d_in[7];
  float* out = (float*)d_out;

  char* ws = (char*)d_ws;
  const size_t kvB = (size_t)NH * NB * NSK * NE * 2;  // 64 MB
  const size_t MB  = (size_t)1 << 20;
  bf16_t* keys     = (bf16_t*)ws;
  bf16_t* valsT    = (bf16_t*)(ws + kvB);
  bf16_t* ctx      = (bf16_t*)(ws + 2 * kvB);
  bf16_t* statesBf = (bf16_t*)(ws + 3 * kvB);              // 16 MB
  bf16_t* WkT      = (bf16_t*)(ws + 3 * kvB + 16 * MB);    // 8 MB
  bf16_t* WvT      = (bf16_t*)(ws + 3 * kvB + 24 * MB);    // 8 MB
  bf16_t* WcT      = (bf16_t*)(ws + 3 * kvB + 32 * MB);    // 1 MB

  // prep: bf16 conversion + weight transposes (one-shot, bandwidth-trivial)
  cvt_states_kernel<<<(NB * NSK * ND) / 1024, 256, 0, stream>>>(states, statesBf);
  tconv_kernel<<<dim3((ND * NE) / 256, NH), 256, 0, stream>>>(Wk, WkT, ND);
  tconv_kernel<<<dim3((ND * NE) / 256, NH), 256, 0, stream>>>(Wv, WvT, ND);
  tconv_kernel<<<dim3((NH * NE * NE) / 256, 1), 256, 0, stream>>>(Wc, WcT, NH * NE);

  dim3 g1(NSK / 128, NH * NB);
  kv_proj_kernel<<<g1, 256, 0, stream>>>(statesBf, WkT, WvT, bk, bv, keys, valsT);

  dim3 g2(NSQ / 16, NB, NH);
  flash_attn_kernel<<<g2, 32, 0, stream>>>(query, keys, valsT, ctx);

  dim3 g3((NB * NSQ) / 128);
  out_proj_kernel<<<g3, 256, 0, stream>>>(ctx, WcT, bc, out);
}
